// BasicModel_77369540870477
// MI455X (gfx1250) — matrix operand order repaired and checked by reading, not compiled, not run on hardware
//
#include <hip/hip_runtime.h>
#include <math.h>

typedef _Float16 v16h __attribute__((ext_vector_type(16)));
typedef _Float16 v8h  __attribute__((ext_vector_type(8)));
typedef float    v8f  __attribute__((ext_vector_type(8)));

#define HID 128
#define G3  384
#define BSZ 32
#define VOC 32000
#define EMB 64
#define PSTR 136   // padded f16 row stride (bank spread for LDS tiles)

// Opaque value pass-through: defeats LICM hoisting (and the resulting spills)
// of loop-invariant WMMA operand loads inside the sequential scan loop.
__device__ __forceinline__ int opaq(int x) {
    asm volatile("" : "+v"(x));
    return x;
}

__device__ __forceinline__ v8f wmma_f16(v16h a, v16h b, v8f c) {
    return __builtin_amdgcn_wmma_f32_16x16x32_f16(false, a, false, b, (short)0, c, false, false);
}

// A operand (16x32 f16) from row-major f16 buffer.
// lane l: row m = mbase + (l&15); half = l>>4
// VGPR0-3: K = kb + half*8 + 0..7 ; VGPR4-7: K = kb + 16 + half*8 + 0..7
__device__ __forceinline__ v16h load_A(const _Float16* base, int stride,
                                       int mbase, int kb, int lane) {
    int m    = mbase + (lane & 15);
    int half = lane >> 4;
    const v8h lo = *(const v8h*)(base + m * stride + kb + half * 8);
    const v8h hi = *(const v8h*)(base + m * stride + kb + 16 + half * 8);
    v16h a;
#pragma unroll
    for (int i = 0; i < 8; ++i) { a[i] = lo[i]; a[i + 8] = hi[i]; }
    return a;
}

// B operand (32x16 f16) where B[k][n] = W[n][k], W row-major [N][stride]:
// lane l: col n; VGPR0-3: K = kb + (l>>4)*8 + 0..7 ; VGPR4-7: K = kb + 16 + (l>>4)*8 + 0..7 (two 16B loads)
__device__ __forceinline__ v16h load_B(const _Float16* W, int stride,
                                       int nrow, int kb, int lane) {
    const _Float16* p = W + nrow * stride + kb + (lane >> 4) * 8;
    const v8h lo = *(const v8h*)(p);
    const v8h hi = *(const v8h*)(p + 16);
    v16h b;
#pragma unroll
    for (int i = 0; i < 8; ++i) { b[i] = lo[i]; b[i + 8] = hi[i]; }
    return b;
}

// ---------------- small helper kernels ----------------

__global__ void convert_f32_to_f16(const float* __restrict__ src,
                                   _Float16* __restrict__ dst, int n) {
    int i = blockIdx.x * blockDim.x + threadIdx.x;
    if (i < n) dst[i] = (_Float16)src[i];
}

__global__ void lengths_kernel(const int* __restrict__ inp, int* __restrict__ lengths) {
    int b = threadIdx.x;
    if (b < BSZ) {
        int cnt = 0;
        for (int l = 0; l < 64; ++l) cnt += (inp[b * 64 + l] != 2) ? 1 : 0;
        lengths[b] = cnt < 63 ? cnt : 63;
    }
}

// xg[l][b][g] = emb[token[b][l]] . Wih[g] + bih[g]    (grid = Lt*32 blocks, 384 thr)
__global__ void gates_kernel(const int* __restrict__ tokens, int tok_stride,
                             const float* __restrict__ emb,
                             const float* __restrict__ Wih,
                             const float* __restrict__ bih,
                             float* __restrict__ xg_t) {
    int l = blockIdx.x >> 5;
    int b = blockIdx.x & 31;
    __shared__ float xs[EMB];
    int tok = tokens[b * tok_stride + l];
    if (threadIdx.x < EMB) xs[threadIdx.x] = emb[tok * EMB + threadIdx.x];
    __syncthreads();
    int g = threadIdx.x;      // 384 threads
    float s = bih[g];
#pragma unroll 8
    for (int e = 0; e < EMB; ++e) s += xs[e] * Wih[g * EMB + e];
    xg_t[(l * BSZ + b) * G3 + g] = s;
}

// dec_h0[b][j] = last[b] . ds_W[j] + ds_b[j]    (grid 16 x 256)
__global__ void dec_h0_kernel(const float* __restrict__ last,
                              const float* __restrict__ dsW,
                              const float* __restrict__ dsb,
                              float* __restrict__ h0) {
    int idx = blockIdx.x * blockDim.x + threadIdx.x;   // 0..4095
    int b = idx >> 7, j = idx & 127;
    float s = dsb[j];
#pragma unroll 8
    for (int k = 0; k < HID; ++k) s += last[b * HID + k] * dsW[j * HID + k];
    h0[b * HID + j] = s;
}

// out[b][0][v] = log(onehot_bos + 1e-9)
__global__ void first_row_kernel(float* __restrict__ out) {
    int idx = blockIdx.x * blockDim.x + threadIdx.x;   // 0..1023999
    int b = idx / VOC, v = idx % VOC;
    float val = (v == 1) ? (1.0f + 1e-9f) : 1e-9f;
    __builtin_nontemporal_store(logf(val), out + b * 64 * VOC + v);
}

// ---------------- GRU scan: one persistent workgroup, WMMA recurrent GEMM ----------------
// 256 threads = 8 waves; wave w: M-tile = (w&1)*16, N-group = (w>>1)*96 (6 tiles of 16).
// Whh (f16) + bhh staged in dynamic LDS once; per-step B operands come from LDS.
__global__ void gru_scan_kernel(const float* __restrict__ xg_t,      // [T][32][384]
                                const _Float16* __restrict__ Whh16,  // [384][128]
                                const float* __restrict__ bhh,       // [384]
                                const float* __restrict__ h0,        // [32][128] or null
                                const int* __restrict__ lengths,     // enc only
                                float* __restrict__ last_state,      // enc only
                                _Float16* __restrict__ dh16,         // dec only [T][32][128]
                                int T) {
    extern __shared__ char smem[];
    float*    h32     = (float*)(smem);                          // 16384 B
    float*    bhh_l   = (float*)(smem + 16384);                  //  1536 B
    _Float16* h16     = (_Float16*)(smem + 17920);               //  8704 B  [32][PSTR]
    _Float16* hg16    = (_Float16*)(smem + 26624);               // 24576 B  [32][384]
    _Float16* whh_l   = (_Float16*)(smem + 51200);               // 104448 B [384][PSTR]

    const int tid  = threadIdx.x;
    const int lane = tid & 31;
    const int wave = tid >> 5;
    const int mt   = (wave & 1) * 16;            // M base (batch rows)
    const int n0   = (wave >> 1) * 96;           // 6 N-tiles of 16

    // stage weights/bias into LDS; init h
    for (int i = tid; i < G3 * HID; i += 256) {
        int n = i >> 7, k = i & 127;
        whh_l[n * PSTR + k] = Whh16[i];
    }
    for (int i = tid; i < G3; i += 256) bhh_l[i] = bhh[i];
#pragma unroll
    for (int i = 0; i < 16; ++i) {
        int idx = i * 256 + tid;                 // 0..4095
        int b = idx >> 7, j = idx & 127;
        float hv = h0 ? h0[idx] : 0.0f;
        h32[idx] = hv;
        h16[b * PSTR + j] = (_Float16)hv;
    }
    __syncthreads();

    for (int t = 0; t < T; ++t) {
        // ---- WMMA phase: hg = h @ Whh^T + bhh ----
        v8f acc[6];
#pragma unroll
        for (int ti = 0; ti < 6; ++ti) {
            float bv = bhh_l[opaq(n0 + ti * 16 + (lane & 15))];
#pragma unroll
            for (int r = 0; r < 8; ++r) acc[ti][r] = bv;
        }
#pragma unroll
        for (int kc = 0; kc < 4; ++kc) {
            int kb = kc * 32;
            v16h a = load_A(h16, PSTR, mt, kb, lane);
#pragma unroll
            for (int ti = 0; ti < 6; ++ti) {
                int nrow = opaq(n0 + ti * 16 + (lane & 15));
                v16h bmat = load_B(whh_l, PSTR, nrow, kb, lane);
                acc[ti] = wmma_f16(a, bmat, acc[ti]);
            }
        }
#pragma unroll
        for (int ti = 0; ti < 6; ++ti) {
            int col = n0 + ti * 16 + (lane & 15);
            int row = mt + (lane >> 4) * 8;
#pragma unroll
            for (int r = 0; r < 8; ++r)
                hg16[(row + r) * G3 + col] = (_Float16)acc[ti][r];
        }
        __syncthreads();

        // ---- gate phase (f32) ----
#pragma unroll
        for (int i = 0; i < 16; ++i) {
            int idx = i * 256 + tid;
            int b = idx >> 7, j = idx & 127;
            const float* xg = xg_t + (t * BSZ + b) * G3;
            float xr = xg[j], xz = xg[j + 128], xn = xg[j + 256];
            float hr = (float)hg16[b * G3 + j];
            float hz = (float)hg16[b * G3 + j + 128];
            float hn = (float)hg16[b * G3 + j + 256];
            float hprev = h32[idx];
            float r = 1.0f / (1.0f + __expf(-(xr + hr)));
            float z = 1.0f / (1.0f + __expf(-(xz + hz)));
            float n = tanhf(xn + r * hn);
            float hnew = (1.0f - z) * n + z * hprev;
            h32[idx] = hnew;
            h16[b * PSTR + j] = (_Float16)hnew;
            if (last_state && lengths[b] == t) last_state[idx] = hnew;
            if (dh16) dh16[(t * BSZ + b) * HID + j] = (_Float16)hnew;
        }
        __syncthreads();
    }
}

// ---------------- logits: [2016,128] x [32000,128]^T, store-bound WMMA ----------------
// grid: x = N block (256 cols, 125 blocks exact), y = M block (32 rows, 63 blocks).
// 256 threads = 8 waves; wave: M-tile = (w&1)*16, 4 N-tiles of 16 at (w>>1)*64.
__global__ void logits_wmma_kernel(const _Float16* __restrict__ dh16,  // [2016][128]
                                   const _Float16* __restrict__ W16,   // [32000][128]
                                   const float* __restrict__ log_b,    // [32000]
                                   float* __restrict__ out) {          // [32][64][32000]
    const int lane  = threadIdx.x & 31;
    const int wave  = threadIdx.x >> 5;
    const int mbase = blockIdx.y * 32 + (wave & 1) * 16;
    const int nbase = blockIdx.x * 256 + (wave >> 1) * 64;

    v8f acc[4];
#pragma unroll
    for (int ti = 0; ti < 4; ++ti) {
        float bv = log_b[nbase + ti * 16 + (lane & 15)];
#pragma unroll
        for (int r = 0; r < 8; ++r) acc[ti][r] = bv;
    }

#pragma unroll 1
    for (int kc = 0; kc < 4; ++kc) {
        int kb = kc * 32;
        v16h a = load_A(dh16, HID, mbase, kb, lane);
#pragma unroll
        for (int ti = 0; ti < 4; ++ti) {
            int col = nbase + ti * 16 + (lane & 15);
            v16h bmat = load_B(W16, HID, col, kb, lane);
            acc[ti] = wmma_f16(a, bmat, acc[ti]);
        }
    }

#pragma unroll
    for (int ti = 0; ti < 4; ++ti) {
        int col = nbase + ti * 16 + (lane & 15);
#pragma unroll
        for (int r = 0; r < 8; ++r) {
            int m = mbase + (lane >> 4) * 8 + r;   // = t*32 + b, t in 0..62
            int tt = m >> 5, bb = m & 31;
            __builtin_nontemporal_store(acc[ti][r],
                                        out + bb * (64 * VOC) + (tt + 1) * VOC + col);
        }
    }
}

// ---------------- host launch ----------------
extern "C" void kernel_launch(void* const* d_in, const int* in_sizes, int n_in,
                              void* d_out, int out_size, void* d_ws, size_t ws_size,
                              hipStream_t stream) {
    const int*   inp     = (const int*)  d_in[0];
    const int*   outtok  = (const int*)  d_in[1];
    const float* emb_inp = (const float*)d_in[2];
    const float* emb_out = (const float*)d_in[3];
    const float* enc_Wih = (const float*)d_in[4];
    const float* enc_Whh = (const float*)d_in[5];
    const float* enc_bih = (const float*)d_in[6];
    const float* enc_bhh = (const float*)d_in[7];
    const float* ds_W    = (const float*)d_in[8];
    const float* ds_b    = (const float*)d_in[9];
    const float* dec_Wih = (const float*)d_in[10];
    const float* dec_Whh = (const float*)d_in[11];
    const float* dec_bih = (const float*)d_in[12];
    const float* dec_bhh = (const float*)d_in[13];
    const float* log_W   = (const float*)d_in[14];
    const float* log_b   = (const float*)d_in[15];
    float* out = (float*)d_out;

    char* ws = (char*)d_ws;
    size_t off = 0;
    auto alloc = [&](size_t bytes) { char* p = ws + off; off += (bytes + 255) & ~size_t(255); return p; };

    _Float16* W16     = (_Float16*)alloc((size_t)VOC * HID * 2);   // 8.2 MB, L2-resident
    _Float16* Whh16e  = (_Float16*)alloc(G3 * HID * 2);
    _Float16* Whh16d  = (_Float16*)alloc(G3 * HID * 2);
    float*    xg_enc  = (float*)   alloc((size_t)64 * BSZ * G3 * 4);
    float*    xg_dec  = (float*)   alloc((size_t)63 * BSZ * G3 * 4);
    int*      lengths = (int*)     alloc(BSZ * 4);
    float*    last_st = (float*)   alloc(BSZ * HID * 4);
    float*    dec_h0  = (float*)   alloc(BSZ * HID * 4);
    _Float16* dh16    = (_Float16*)alloc((size_t)63 * BSZ * HID * 2);

    // precision conversion (f16 weights for WMMA B-operands)
    {
        int n = VOC * HID;
        convert_f32_to_f16<<<(n + 255) / 256, 256, 0, stream>>>(log_W, W16, n);
        n = G3 * HID;
        convert_f32_to_f16<<<(n + 255) / 256, 256, 0, stream>>>(enc_Whh, Whh16e, n);
        convert_f32_to_f16<<<(n + 255) / 256, 256, 0, stream>>>(dec_Whh, Whh16d, n);
    }

    lengths_kernel<<<1, 32, 0, stream>>>(inp, lengths);

    const size_t scan_smem = 51200 + (size_t)G3 * PSTR * 2;   // ~152 KB dynamic LDS

    // encoder
    gates_kernel<<<64 * BSZ, G3, 0, stream>>>(inp, 64, emb_inp, enc_Wih, enc_bih, xg_enc);
    gru_scan_kernel<<<1, 256, scan_smem, stream>>>(xg_enc, Whh16e, enc_bhh,
                                                   nullptr, lengths, last_st, nullptr, 64);

    // bridge + decoder
    dec_h0_kernel<<<16, 256, 0, stream>>>(last_st, ds_W, ds_b, dec_h0);
    gates_kernel<<<63 * BSZ, G3, 0, stream>>>(outtok, 64, emb_out, dec_Wih, dec_bih, xg_dec);
    gru_scan_kernel<<<1, 256, scan_smem, stream>>>(xg_dec, Whh16d, dec_bhh,
                                                   dec_h0, nullptr, nullptr, dh16, 63);

    // logits (dominant, store-bound WMMA GEMM), 125x63 blocks, no remainder
    dim3 grid(125, 63);
    logits_wmma_kernel<<<grid, 256, 0, stream>>>(dh16, W16, log_b, out);

    // t=0 row: log(onehot(BOS)+1e-9)
    first_row_kernel<<<(BSZ * VOC) / 256, 256, 0, stream>>>(out);
}
